// DotProductAttentionWithRPR_21449066676706
// MI455X (gfx1250) — matrix operand order repaired and checked by reading, not compiled, not run on hardware
//
#include <hip/hip_runtime.h>
#include <hip/hip_bf16.h>

// Attention with relative positional representations for MI455X (gfx1250).
// B=32, TQ=TK=1024, D=64, f32 in/out, bf16 WMMA internally (f32 accumulate).
//
// Grid: 128 workgroups = 64 q-tiles x 2 batch-groups. Block: 512 threads (16 wave32).
// Wave w plays two roles per 32-key step:
//   content role: batch cb=w   -> Q[b] @ K[b]^T  and  P[b] @ V[b]        (M = query rows)
//   pos role:     query  pm=w  -> Qall[:,q,:] @ pos_k[q]^T and
//                                 Pall[:,q,:] @ pos_v[q]                 (M = batch rows)
// Scores are combined, scaled, masked in LDS; online softmax per (b,m) row.

#define TQd 1024
#define TKd 1024
#define DIMd 64

typedef __attribute__((ext_vector_type(16))) __bf16 bf16x16;
typedef __attribute__((ext_vector_type(8)))  float  f32x8;

static __device__ __forceinline__ __bf16 f2bf(float x) { return (__bf16)x; }

// A-fragment (16x32 bf16) k index for VGPR pair p (0..7), lane-half hi:
// lanes 0-15: V0..V7 -> K 0..15 ; lanes 16-31: K 16..31 (contiguous, matches B)
static __device__ __forceinline__ int amap_k(int p, int hi) {
  return (p << 1) + (hi << 4);
}

static __device__ __forceinline__ f32x8 wmma_acc(bf16x16 a, bf16x16 b, f32x8 c) {
  return __builtin_amdgcn_wmma_f32_16x16x32_bf16(false, a, false, b, (short)0, c,
                                                 false, false);
}

// Load A fragment from f32 row (row = M index already applied), d-range base d0.
static __device__ __forceinline__ bf16x16 load_A_f32(const float* rowbase, int d0, int hi) {
  bf16x16 a;
#pragma unroll
  for (int p = 0; p < 8; ++p) {
    int k = d0 + amap_k(p, hi);
    float2 f = *(const float2*)(rowbase + k);   // k even -> 8B aligned
    a[2 * p]     = f2bf(f.x);
    a[2 * p + 1] = f2bf(f.y);
  }
  return a;
}

// Load B fragment (32x16) where the reduction dim is contiguous f32 in memory.
// Element e -> rowptr[d0 + e]; d0 already includes hi*16. 4x b128 loads.
static __device__ __forceinline__ bf16x16 load_B_f32rows(const float* rowptr, int d0) {
  bf16x16 b;
#pragma unroll
  for (int q = 0; q < 4; ++q) {
    float4 f = *(const float4*)(rowptr + d0 + q * 4);  // 16B aligned (d0 mult of 16)
    b[4 * q + 0] = f2bf(f.x);
    b[4 * q + 1] = f2bf(f.y);
    b[4 * q + 2] = f2bf(f.z);
    b[4 * q + 3] = f2bf(f.w);
  }
  return b;
}

// LDS layout (bytes):
//   [0      , 65536 )  vtV : V^T tile,   bf16 [16 b][64 d][32 k]
//   [65536  , 131072)  vtP : posV^T tile bf16 [16 m][64 d][32 k]
//   [131072 , 167936)  S   : scores f32 [16 b][16 m][36]  (padded stride)
//   [167936 , 184320)  P   : probs bf16 [16 b][16 m][32]
//   [184320 , 185344)  FAC : per-row rescale factor / 1/l, f32 [256]
//   epilogue reuses [0,65536) as O-combine buffer f32 [16 b][16 m][64 d]
#define SMEM_BYTES 185344

__launch_bounds__(512, 1)
__global__ void attn_rpr_wmma_kernel(const float* __restrict__ Q,
                                     const float* __restrict__ K,
                                     const float* __restrict__ V,
                                     const float* __restrict__ PK,
                                     const float* __restrict__ PV,
                                     const int*   __restrict__ VL,
                                     float* __restrict__ out) {
  extern __shared__ __align__(128) char smem[];
  __bf16* vtV = (__bf16*)(smem);
  __bf16* vtP = (__bf16*)(smem + 65536);
  float*  S   = (float*)(smem + 131072);
  __bf16* Pb  = (__bf16*)(smem + 167936);
  float*  FAC = (float*)(smem + 184320);
  float*  OB  = (float*)(smem);  // epilogue only

  const int tid  = threadIdx.x;
  const int wv   = tid >> 5;
  const int lane = tid & 31;
  const int ln   = lane & 15;
  const int hi   = lane >> 4;

  const int qt = (int)blockIdx.x >> 1;
  const int bg = (int)blockIdx.x & 1;
  const int q0 = qt * 16;
  const int cb = wv;               // content-role batch within group
  const int pm = wv;               // pos-role query row within tile
  const int bglob = bg * 16 + cb;
  const int qm = q0 + pm;

  // valid_lens for pos-role score rows: row v of C fragment is batch (v + 8*hi)
  int vl[8];
#pragma unroll
  for (int v = 0; v < 8; ++v) vl[v] = VL[bg * 16 + v + 8 * hi];

  // Loop-invariant A fragments (Q in two d-halves, bf16)
  const float* qcrow = Q + ((size_t)bglob * TQd + q0 + ln) * DIMd;  // row = query m
  bf16x16 aQc0 = load_A_f32(qcrow, 0, hi);
  bf16x16 aQc1 = load_A_f32(qcrow, 32, hi);
  const float* qprow = Q + ((size_t)(bg * 16 + ln) * TQd + qm) * DIMd;  // row = batch
  bf16x16 aQp0 = load_A_f32(qprow, 0, hi);
  bf16x16 aQp1 = load_A_f32(qprow, 32, hi);

  f32x8 Oc[4], Op[4];
#pragma unroll
  for (int dt = 0; dt < 4; ++dt) {
#pragma unroll
    for (int v = 0; v < 8; ++v) { Oc[dt][v] = 0.f; Op[dt][v] = 0.f; }
  }

  float rm = -3.0e38f, lsum = 0.0f;  // online-softmax row state (tid < 256)
  const int srow = tid;              // row = b*16 + m for tid < 256

  const float* Vbase  = V  + ((size_t)bg * 16) * TKd * DIMd;
  const float* PVbase = PV + ((size_t)q0) * (size_t)TKd * DIMd;
  const int dblk = tid & 15;   // staging: d block
  const int rb   = tid >> 4;   // staging: row base 0..31

#pragma unroll 1
  for (int kt = 0; kt < TKd / 32; ++kt) {
    const int kbase = kt * 32;

    // ---- (a) stage V^T and posV^T tiles into LDS (f32 -> bf16, transposed) ----
#pragma unroll 4
    for (int i = 0; i < 16; ++i) {
      int rid = i * 32 + rb;          // 0..511
      int b = rid >> 5, kk = rid & 31;
      size_t go = ((size_t)b * TKd + kbase + kk) * DIMd + dblk * 4;
      float4 fv = *(const float4*)(Vbase + go);
      float4 fp = *(const float4*)(PVbase + go);
      int lo = (b * 64 + dblk * 4) * 32 + kk;   // [slab][d][k]
      vtV[lo]      = f2bf(fv.x); vtV[lo + 32] = f2bf(fv.y);
      vtV[lo + 64] = f2bf(fv.z); vtV[lo + 96] = f2bf(fv.w);
      vtP[lo]      = f2bf(fp.x); vtP[lo + 32] = f2bf(fp.y);
      vtP[lo + 64] = f2bf(fp.z); vtP[lo + 96] = f2bf(fp.w);
    }

    // ---- (b) content scores: S_cont = Q[b] @ K[b]^T (per wave, its batch) ----
#pragma unroll
    for (int nt = 0; nt < 2; ++nt) {
      const float* krow = K + ((size_t)bglob * TKd + kbase + nt * 16 + ln) * DIMd;
      __builtin_prefetch(krow + 32 * DIMd, 0, 0);  // next step's K row
      f32x8 c = {0.f, 0.f, 0.f, 0.f, 0.f, 0.f, 0.f, 0.f};
      c = wmma_acc(aQc0, load_B_f32rows(krow, hi * 16), c);
      c = wmma_acc(aQc1, load_B_f32rows(krow, 32 + hi * 16), c);
#pragma unroll
      for (int v = 0; v < 8; ++v)
        S[(cb * 16 + v + 8 * hi) * 36 + nt * 16 + ln] = c[v];
    }
    __syncthreads();

    // ---- (d) pos scores + combine + scale + mask (per wave, its query row) ----
#pragma unroll
    for (int nt = 0; nt < 2; ++nt) {
      const float* prow = PK + ((size_t)qm * TKd + kbase + nt * 16 + ln) * DIMd;
      __builtin_prefetch(prow + 32 * DIMd, 0, 0);  // next step's pos_k row
      f32x8 c = {0.f, 0.f, 0.f, 0.f, 0.f, 0.f, 0.f, 0.f};
      c = wmma_acc(aQp0, load_B_f32rows(prow, hi * 16), c);
      c = wmma_acc(aQp1, load_B_f32rows(prow, 32 + hi * 16), c);
      const int kg = kbase + nt * 16 + ln;  // global key index (lane-invariant n)
#pragma unroll
      for (int v = 0; v < 8; ++v) {
        int idx = ((v + 8 * hi) * 16 + pm) * 36 + nt * 16 + ln;
        float s = (S[idx] + c[v]) * 0.125f;          // 1/sqrt(64)
        S[idx] = (kg < vl[v]) ? s : -1.0e6f;         // reference masking
      }
    }
    __syncthreads();

    // ---- (f) online softmax, one (b,m) row per thread ----
    if (tid < 256) {
      float* row = S + srow * 36;
      float mx = rm;
#pragma unroll
      for (int j = 0; j < 32; ++j) mx = fmaxf(mx, row[j]);
      float fac = __expf(rm - mx);
      float sum = 0.f;
      __bf16* prow_ = Pb + srow * 32;
#pragma unroll
      for (int j = 0; j < 32; ++j) {
        float p = __expf(row[j] - mx);
        sum += p;
        prow_[j] = f2bf(p);
      }
      lsum = lsum * fac + sum;
      rm = mx;
      FAC[srow] = fac;
    }
    __syncthreads();

    // ---- (h) PV: rescale accumulators, then both roles' matmuls ----
    float fc[8], fpz[8];
#pragma unroll
    for (int v = 0; v < 8; ++v) {
      fc[v]  = FAC[cb * 16 + v + 8 * hi];        // content rows are m
      fpz[v] = FAC[(v + 8 * hi) * 16 + pm];      // pos rows are b
    }
#pragma unroll
    for (int dt = 0; dt < 4; ++dt) {
#pragma unroll
      for (int v = 0; v < 8; ++v) { Oc[dt][v] *= fc[v]; Op[dt][v] *= fpz[v]; }
    }

    bf16x16 aPc, aPp;
#pragma unroll
    for (int p = 0; p < 8; ++p) {
      int k = amap_k(p, hi);
      const __bf16* pc = Pb + (cb * 16 + ln) * 32 + k;  // rows = m
      aPc[2 * p] = pc[0]; aPc[2 * p + 1] = pc[1];
      const __bf16* pp = Pb + (ln * 16 + pm) * 32 + k;  // rows = b
      aPp[2 * p] = pp[0]; aPp[2 * p + 1] = pp[1];
    }
#pragma unroll
    for (int dt = 0; dt < 4; ++dt) {
      const __bf16* vb = vtV + (cb * 64 + dt * 16 + ln) * 32 + hi * 16;
      Oc[dt] = wmma_acc(aPc, *(const bf16x16*)vb, Oc[dt]);
      const __bf16* pb = vtP + (pm * 64 + dt * 16 + ln) * 32 + hi * 16;
      Op[dt] = wmma_acc(aPp, *(const bf16x16*)pb, Op[dt]);
    }
    __syncthreads();
  }

  // ---- epilogue: combine content + pos accumulators, normalize, store ----
#pragma unroll
  for (int dt = 0; dt < 4; ++dt) {
#pragma unroll
    for (int v = 0; v < 8; ++v)
      OB[(cb * 16 + v + 8 * hi) * 64 + dt * 16 + ln] = Oc[dt][v];
  }
  __syncthreads();
#pragma unroll
  for (int dt = 0; dt < 4; ++dt) {
#pragma unroll
    for (int v = 0; v < 8; ++v)
      OB[((v + 8 * hi) * 16 + pm) * 64 + dt * 16 + ln] += Op[dt][v];
  }
  if (tid < 256) FAC[srow] = 1.0f / lsum;
  __syncthreads();
  {
    int row = tid >> 1, half = tid & 1;   // row = b*16 + m
    int b = row >> 4, m = row & 15;
    float li = FAC[row];
    const float* src = OB + row * 64 + half * 32;
    float* dst = out + (((size_t)(bg * 16 + b)) * TQd + q0 + m) * DIMd + half * 32;
#pragma unroll
    for (int j = 0; j < 8; ++j) {
      float4 f = *(const float4*)(src + j * 4);
      f.x *= li; f.y *= li; f.z *= li; f.w *= li;
      *(float4*)(dst + j * 4) = f;
    }
  }
}

extern "C" void kernel_launch(void* const* d_in, const int* in_sizes, int n_in,
                              void* d_out, int out_size, void* d_ws, size_t ws_size,
                              hipStream_t stream) {
  (void)in_sizes; (void)n_in; (void)out_size; (void)d_ws; (void)ws_size;
  const float* Q  = (const float*)d_in[0];
  const float* K  = (const float*)d_in[1];
  const float* V  = (const float*)d_in[2];
  const float* PK = (const float*)d_in[3];
  const float* PV = (const float*)d_in[4];
  const int*   VL = (const int*)d_in[5];
  float* out = (float*)d_out;

  dim3 grid(128);    // 64 q-tiles x 2 batch-groups
  dim3 block(512);   // 16 wave32
  attn_rpr_wmma_kernel<<<grid, block, SMEM_BYTES, stream>>>(Q, K, V, PK, PV, VL, out);
}
